// DFlashAttentionV8_72292889526384
// MI455X (gfx1250) — matrix operand order repaired and checked by reading, not compiled, not run on hardware
//
#include <hip/hip_runtime.h>
#include <hip/hip_bf16.h>
#include <math.h>

#define B_    8
#define K_    64
#define H_    2048
#define NH_   16
#define NKV_  4
#define HD_   128
#define CTX_  4096
#define KVLEN_ 4160          // CTX_ + K_
#define GROUPS_ 4            // NH_/NKV_
#define EPS_  1e-6f
#define SCALE_ 0.08838834764831845f  // 1/sqrt(128)
#define PARTS_ 5
#define TPP_   26            // 32-wide kv tiles per partition (5*26*32 = 4160)

typedef __attribute__((ext_vector_type(16))) _Float16 v16h;
typedef __attribute__((ext_vector_type(8)))  _Float16 v8h;
typedef __attribute__((ext_vector_type(4)))  _Float16 v4h;
typedef __attribute__((ext_vector_type(8)))  float    v8f;
typedef __attribute__((ext_vector_type(4)))  int      v4i;
typedef __attribute__((address_space(1))) v4i as1_v4i;
typedef __attribute__((address_space(3))) v4i as3_v4i;

// ---- gfx1250 async global->LDS copy (probe via __has_builtin) ----
#if defined(__gfx1250__) && __has_builtin(__builtin_amdgcn_global_load_async_to_lds_b128)
#define HAVE_ASYNC_LDS 1
#else
#define HAVE_ASYNC_LDS 0
#endif

__device__ __forceinline__ void copy16(const void* g, void* l) {
#if HAVE_ASYNC_LDS
  __builtin_amdgcn_global_load_async_to_lds_b128(
      (as1_v4i*)(void*)g, (as3_v4i*)l, 0, 0);
#else
  *(uint4*)l = *(const uint4*)g;
#endif
}

__device__ __forceinline__ void wait_async0() {
#if HAVE_ASYNC_LDS
#if __has_builtin(__builtin_amdgcn_s_wait_asynccnt)
  __builtin_amdgcn_s_wait_asynccnt(0);
#else
  asm volatile("s_wait_asynccnt 0x0" ::: "memory");
#endif
#endif
}

// ---------------- workspace layout (bytes, all 256B aligned) ----------------
#define OFF_QRAW 0ull                                   // 512*2048 f32
#define OFF_KRAW (OFF_QRAW + 512ull*2048*4)             // 512*512  f32
#define OFF_VRAW (OFF_KRAW + 512ull*512*4)              // 512*512  f32
#define OFF_QH   (OFF_VRAW + 512ull*512*4)              // 512*2048 f16
#define OFF_KH   (OFF_QH   + 512ull*2048*2)             // B*NKV*KVLEN*HD f16
#define OFF_VTH  (OFF_KH   + (unsigned long long)B_*NKV_*KVLEN_*HD_*2)
#define OFF_ATTN (OFF_VTH  + (unsigned long long)B_*NKV_*KVLEN_*HD_*2) // 512*2048 f32
#define OFF_OPART (OFF_ATTN + 512ull*2048*4)            // PARTS*B*NH*K*HD f32 (20MB)
#define OFF_ML   (OFF_OPART + (unsigned long long)PARTS_*B_*NH_*K_*HD_*4)
// total ~102 MB

// ---------------------------------------------------------------------------
// WMMA fragment loaders (wave32): lane half h holds k = 16h..16h+15 (contiguous order).
__device__ __forceinline__ v16h load_a_frag(const _Float16* base, int ld) {
  int lane = threadIdx.x & 31;
  int m  = lane & 15;
  int k0 = (lane >> 4) * 16;
  return *(const v16h*)(base + m * ld + k0);
}
__device__ __forceinline__ v16h load_bT_frag(const _Float16* brows, int ld) {
  int lane = threadIdx.x & 31;
  int n  = lane & 15;
  int k0 = (lane >> 4) * 16;
  return *(const v16h*)(brows + n * ld + k0);
}
__device__ __forceinline__ v8f wmma_f16(v16h a, v16h b, v8f c) {
  return __builtin_amdgcn_wmma_f32_16x16x32_f16(false, a, false, b, (short)0, c,
                                                false, false);
}

// ---------------------------------------------------------------------------
// C = A(MxKd f32) @ W(NxKd f32)^T -> f32 MxN. 128 thr (4 waves), 64x64 tile.
__global__ void gemm_xwT(const float* __restrict__ A, const float* __restrict__ W,
                         float* __restrict__ C, int M, int N, int Kd) {
  __shared__ __align__(32) _Float16 At[64][32];
  __shared__ __align__(32) _Float16 Bt[64][32];
  const int m0 = blockIdx.y * 64, n0 = blockIdx.x * 64;
  const int w = threadIdx.x >> 5, lane = threadIdx.x & 31;
  const int lo = lane & 15, hi = lane >> 4;
  const int rr = threadIdx.x >> 3;        // 0..15
  const int c4 = (threadIdx.x & 7) * 4;   // 0..28

  v8f acc[4];
#pragma unroll
  for (int t = 0; t < 4; ++t)
#pragma unroll
    for (int r = 0; r < 8; ++r) acc[t][r] = 0.0f;

  for (int kb = 0; kb < Kd; kb += 32) {
#pragma unroll
    for (int i = 0; i < 4; ++i) {
      int r = rr + 16 * i;
      float4 av = *(const float4*)&A[(size_t)(m0 + r) * Kd + kb + c4];
      float4 wv = *(const float4*)&W[(size_t)(n0 + r) * Kd + kb + c4];
      v4h ah, wh;
      ah[0] = (_Float16)av.x; ah[1] = (_Float16)av.y;
      ah[2] = (_Float16)av.z; ah[3] = (_Float16)av.w;
      wh[0] = (_Float16)wv.x; wh[1] = (_Float16)wv.y;
      wh[2] = (_Float16)wv.z; wh[3] = (_Float16)wv.w;
      *(v4h*)&At[r][c4] = ah;
      *(v4h*)&Bt[r][c4] = wh;
    }
    __syncthreads();
    v16h a = load_a_frag(&At[16 * w][0], 32);
#pragma unroll
    for (int t = 0; t < 4; ++t) {
      v16h b = load_bT_frag(&Bt[16 * t][0], 32);
      acc[t] = wmma_f16(a, b, acc[t]);
    }
    __syncthreads();
  }
#pragma unroll
  for (int t = 0; t < 4; ++t)
#pragma unroll
    for (int r = 0; r < 8; ++r)
      C[(size_t)(m0 + 16 * w + r + 8 * hi) * N + n0 + 16 * t + lo] = acc[t][r];
}

// ---------------------------------------------------------------------------
// RMSNorm + RoPE + f16 scatter for freshly projected Q / K_new / V_new.
__global__ void fuse_norm_rope_new(const float* __restrict__ qraw,
                                   const float* __restrict__ kraw,
                                   const float* __restrict__ vraw,
                                   const float* __restrict__ cosT,
                                   const float* __restrict__ sinT,
                                   const float* __restrict__ qw,
                                   const float* __restrict__ kw,
                                   _Float16* __restrict__ qh,
                                   _Float16* __restrict__ kh,
                                   _Float16* __restrict__ vth) {
  __shared__ float red[128];
  __shared__ float ybuf[128];
  const int d = threadIdx.x;
  const unsigned vblk = blockIdx.x;

  if (vblk < 8192u) {  // Q: 512 tokens * 16 heads
    int m = vblk / NH_, h = vblk % NH_;
    int b = m >> 6, kq = m & 63, pos = CTX_ + kq;
    float x = qraw[(size_t)m * H_ + h * HD_ + d];
    red[d] = x * x; __syncthreads();
    for (int s = 64; s > 0; s >>= 1) { if (d < s) red[d] += red[d + s]; __syncthreads(); }
    float y = x * rsqrtf(red[0] / HD_ + EPS_) * qw[d];
    __syncthreads();
    ybuf[d] = y; __syncthreads();
    float rot = (d < 64) ? -ybuf[d + 64] : ybuf[d - 64];
    size_t ci = ((size_t)b * KVLEN_ + pos) * HD_ + d;
    float o = y * cosT[ci] + rot * sinT[ci];
    qh[(((size_t)b * NH_ + h) * K_ + kq) * HD_ + d] = (_Float16)o;
  } else if (vblk < 10240u) {  // K_new
    int v2 = vblk - 8192;
    int m = v2 / NKV_, g = v2 % NKV_;
    int b = m >> 6, kq = m & 63, pos = CTX_ + kq;
    float x = kraw[(size_t)m * 512 + g * HD_ + d];
    red[d] = x * x; __syncthreads();
    for (int s = 64; s > 0; s >>= 1) { if (d < s) red[d] += red[d + s]; __syncthreads(); }
    float y = x * rsqrtf(red[0] / HD_ + EPS_) * kw[d];
    __syncthreads();
    ybuf[d] = y; __syncthreads();
    float rot = (d < 64) ? -ybuf[d + 64] : ybuf[d - 64];
    size_t ci = ((size_t)b * KVLEN_ + pos) * HD_ + d;
    float o = y * cosT[ci] + rot * sinT[ci];
    kh[(((size_t)b * NKV_ + g) * KVLEN_ + pos) * HD_ + d] = (_Float16)o;
  } else {  // V_new -> transposed [HD][kv]
    int v3 = vblk - 10240;
    int m = v3 / NKV_, g = v3 % NKV_;
    int b = m >> 6, kq = m & 63;
    float x = vraw[(size_t)m * 512 + g * HD_ + d];
    vth[(((size_t)b * NKV_ + g) * HD_ + d) * KVLEN_ + CTX_ + kq] = (_Float16)x;
  }
}

// ---------------------------------------------------------------------------
__global__ void ctx_convert(const float* __restrict__ ctxk,
                            const float* __restrict__ ctxv,
                            const float* __restrict__ cosT,
                            const float* __restrict__ sinT,
                            const float* __restrict__ kw,
                            _Float16* __restrict__ kh,
                            _Float16* __restrict__ vth) {
  __shared__ float red[128];
  __shared__ float ybuf[128];
  const int d = threadIdx.x;
  unsigned v = blockIdx.x;
  int g = v % NKV_;
  int t = (v / NKV_) % CTX_;
  int b = v / (NKV_ * CTX_);
  size_t src = (((size_t)b * CTX_ + t) * NKV_ + g) * HD_ + d;

  float x = ctxk[src];
  red[d] = x * x; __syncthreads();
  for (int s = 64; s > 0; s >>= 1) { if (d < s) red[d] += red[d + s]; __syncthreads(); }
  float y = x * rsqrtf(red[0] / HD_ + EPS_) * kw[d];
  __syncthreads();
  ybuf[d] = y; __syncthreads();
  float rot = (d < 64) ? -ybuf[d + 64] : ybuf[d - 64];
  size_t ci = ((size_t)b * KVLEN_ + t) * HD_ + d;
  float o = y * cosT[ci] + rot * sinT[ci];
  kh[(((size_t)b * NKV_ + g) * KVLEN_ + t) * HD_ + d] = (_Float16)o;

  float xv = ctxv[src];
  vth[(((size_t)b * NKV_ + g) * HD_ + d) * KVLEN_ + t] = (_Float16)xv;
}

// ---------------------------------------------------------------------------
// Stage one 32-kv tile of K ([32][128] f16, contiguous 8KB) and Vt
// ([128][32] f16, 64B rows) into LDS (async on gfx1250, 16B chunks).
__device__ __forceinline__ void stage_tile(_Float16* Ktd, _Float16* Vtd,
                                           const _Float16* ksrc,
                                           const _Float16* vsrc, int tid) {
#pragma unroll
  for (int i = 0; i < 4; ++i) {
    int ch = tid + 128 * i;                           // 0..511, 16B each
    copy16((const char*)ksrc + ch * 16, (char*)Ktd + ch * 16);
    int row = ch >> 2, off = (ch & 3) * 16;
    copy16((const char*)(vsrc + (size_t)row * KVLEN_) + off, (char*)Vtd + ch * 16);
  }
}

// ---------------------------------------------------------------------------
// Flash attention partition: block=(b,h)xPART, 4 waves x 16 q rows, online
// softmax, K/V tiles double-buffered in LDS, unnormalized partial O + (m,l).
__global__ void flash_attn_part(const _Float16* __restrict__ qh,
                                const _Float16* __restrict__ kh,
                                const _Float16* __restrict__ vth,
                                const float* __restrict__ mask,
                                float* __restrict__ Opart,
                                float* __restrict__ mlpart) {
  __shared__ __align__(32) _Float16 Kt[2][32][128];
  __shared__ __align__(32) _Float16 Vt[2][128][32];
  __shared__ __align__(32) float    Slds[4][16][32];
  __shared__ __align__(32) _Float16 Plds[4][16][32];
  __shared__ float Alds[4][16];

  const int b = blockIdx.x / NH_, h = blockIdx.x % NH_, g = h / GROUPS_;
  const int p = blockIdx.y;
  const int tid = threadIdx.x;
  const int w = tid >> 5, lane = tid & 31, lo = lane & 15, hi = lane >> 4;
  const int j0 = p * (TPP_ * 32);

  const _Float16* qbase = qh + (((size_t)b * NH_ + h) * K_ + 16 * w) * HD_;
  const _Float16* kbase = kh + ((size_t)b * NKV_ + g) * (size_t)KVLEN_ * HD_;
  const _Float16* vbase = vth + ((size_t)b * NKV_ + g) * (size_t)HD_ * KVLEN_;

  v16h qf[4];
#pragma unroll
  for (int kc = 0; kc < 4; ++kc) qf[kc] = load_a_frag(qbase + 32 * kc, HD_);

  v8f o[8];
#pragma unroll
  for (int c = 0; c < 8; ++c)
#pragma unroll
    for (int r = 0; r < 8; ++r) o[c][r] = 0.0f;

  float m_i = -INFINITY, l_i = 0.0f;

  stage_tile(&Kt[0][0][0], &Vt[0][0][0], kbase + (size_t)j0 * HD_, vbase + j0, tid);
  wait_async0();
  __syncthreads();

  for (int jt = 0; jt < TPP_; ++jt) {
    const int bi = jt & 1;
    const int j = j0 + jt * 32;
    if (jt + 1 < TPP_) {
      int jn = j + 32;
      stage_tile(&Kt[bi ^ 1][0][0], &Vt[bi ^ 1][0][0],
                 kbase + (size_t)jn * HD_, vbase + jn, tid);
    }

    // ---- S = Q @ K^T from LDS tile ----
#pragma unroll
    for (int t = 0; t < 2; ++t) {
      v8f s;
#pragma unroll
      for (int r = 0; r < 8; ++r) s[r] = 0.0f;
#pragma unroll
      for (int kc = 0; kc < 4; ++kc) {
        v16h kf = load_bT_frag(&Kt[bi][16 * t][32 * kc], 128);
        s = wmma_f16(qf[kc], kf, s);
      }
#pragma unroll
      for (int r = 0; r < 8; ++r)
        Slds[w][r + 8 * hi][16 * t + lo] = s[r];
    }
    __syncthreads();

    // ---- online softmax: lanes 0..15 own one query row; keep scores in LDS ----
    if (lane < 16) {
      int q = 16 * w + lane;
      const float* mrow = mask + ((size_t)b * K_ + q) * KVLEN_ + j;
      float mx = -INFINITY;
#pragma unroll
      for (int c = 0; c < 32; ++c) {
        float fv = Slds[w][lane][c] * SCALE_ + mrow[c];
        Slds[w][lane][c] = fv;
        mx = fmaxf(mx, fv);
      }
      float m_new = fmaxf(m_i, mx);
      float al = __expf(m_i - m_new);
      float sum = 0.0f;
#pragma unroll
      for (int c = 0; c < 32; ++c) {
        float pe = __expf(Slds[w][lane][c] - m_new);
        sum += pe;
        Plds[w][lane][c] = (_Float16)pe;
      }
      l_i = l_i * al + sum;
      m_i = m_new;
      Alds[w][lane] = al;
    }
    __syncthreads();

    // ---- O = O*alpha + P @ V ----
    float ar[8];
#pragma unroll
    for (int r = 0; r < 8; ++r) ar[r] = Alds[w][r + 8 * hi];
    v16h pa = load_a_frag(&Plds[w][0][0], 32);
#pragma unroll
    for (int c = 0; c < 8; ++c) {
#pragma unroll
      for (int r = 0; r < 8; ++r) o[c][r] *= ar[r];
      v16h vf = load_bT_frag(&Vt[bi][16 * c][0], 32);
      o[c] = wmma_f16(pa, vf, o[c]);
    }
    wait_async0();
    __syncthreads();
  }

  // ---- store partial (unnormalized O, and per-row m,l) ----
  if (lane < 16) {
    int q = 16 * w + lane;
    size_t mi = ((((size_t)p * B_ + b) * NH_ + h) * K_ + q) * 2;
    mlpart[mi] = m_i;
    mlpart[mi + 1] = l_i;
  }
#pragma unroll
  for (int c = 0; c < 8; ++c)
#pragma unroll
    for (int r = 0; r < 8; ++r) {
      int q = 16 * w + r + 8 * hi;
      Opart[((((size_t)p * B_ + b) * NH_ + h) * K_ + q) * HD_ + 16 * c + lo] = o[c][r];
    }
}

// ---------------------------------------------------------------------------
// Merge KV partitions: out = sum_p e^{m_p-m}*O_p / sum_p e^{m_p-m}*l_p.
__global__ void attn_combine(const float* __restrict__ Opart,
                             const float* __restrict__ mlpart,
                             float* __restrict__ attn_out) {
  const int d = threadIdx.x;
  unsigned v = blockIdx.x;                 // b*NH*K blocks
  int q = v % K_;
  int h = (v / K_) % NH_;
  int b = v / (K_ * NH_);

  float mv[PARTS_], lv[PARTS_];
  float m_tot = -INFINITY;
#pragma unroll
  for (int p = 0; p < PARTS_; ++p) {
    size_t mi = ((((size_t)p * B_ + b) * NH_ + h) * K_ + q) * 2;
    mv[p] = mlpart[mi];
    lv[p] = mlpart[mi + 1];
    m_tot = fmaxf(m_tot, mv[p]);
  }
  float denom = 0.0f, acc = 0.0f;
#pragma unroll
  for (int p = 0; p < PARTS_; ++p) {
    float sc = __expf(mv[p] - m_tot);
    denom += sc * lv[p];
    acc += sc * Opart[((((size_t)p * B_ + b) * NH_ + h) * K_ + q) * HD_ + d];
  }
  attn_out[((size_t)b * K_ + q) * (NH_ * HD_) + h * HD_ + d] = acc / denom;
}

// ---------------------------------------------------------------------------
extern "C" void kernel_launch(void* const* d_in, const int* in_sizes, int n_in,
                              void* d_out, int out_size, void* d_ws, size_t ws_size,
                              hipStream_t stream) {
  (void)in_sizes; (void)n_in; (void)out_size; (void)ws_size;
  const float* hidden = (const float*)d_in[0];
  const float* ctx_k  = (const float*)d_in[1];
  const float* ctx_v  = (const float*)d_in[2];
  const float* amask  = (const float*)d_in[3];
  const float* cosT   = (const float*)d_in[4];
  const float* sinT   = (const float*)d_in[5];
  const float* wq     = (const float*)d_in[6];
  const float* wk     = (const float*)d_in[7];
  const float* wv     = (const float*)d_in[8];
  const float* wo     = (const float*)d_in[9];
  const float* qnw    = (const float*)d_in[10];
  const float* knw    = (const float*)d_in[11];

  char* ws = (char*)d_ws;  // needs ~102 MB
  float*    qraw  = (float*)(ws + OFF_QRAW);
  float*    kraw  = (float*)(ws + OFF_KRAW);
  float*    vraw  = (float*)(ws + OFF_VRAW);
  _Float16* qhp   = (_Float16*)(ws + OFF_QH);
  _Float16* khp   = (_Float16*)(ws + OFF_KH);
  _Float16* vthp  = (_Float16*)(ws + OFF_VTH);
  float*    attnw = (float*)(ws + OFF_ATTN);
  float*    opart = (float*)(ws + OFF_OPART);
  float*    mlp   = (float*)(ws + OFF_ML);

  dim3 blk(128);
  gemm_xwT<<<dim3(2048 / 64, 512 / 64), blk, 0, stream>>>(hidden, wq, qraw, 512, 2048, 2048);
  gemm_xwT<<<dim3(512 / 64, 512 / 64), blk, 0, stream>>>(hidden, wk, kraw, 512, 512, 2048);
  gemm_xwT<<<dim3(512 / 64, 512 / 64), blk, 0, stream>>>(hidden, wv, vraw, 512, 512, 2048);
  fuse_norm_rope_new<<<12288, blk, 0, stream>>>(qraw, kraw, vraw, cosT, sinT, qnw, knw,
                                                qhp, khp, vthp);
  ctx_convert<<<B_ * CTX_ * NKV_, blk, 0, stream>>>(ctx_k, ctx_v, cosT, sinT, knw,
                                                    khp, vthp);
  flash_attn_part<<<dim3(B_ * NH_, PARTS_), blk, 0, stream>>>(qhp, khp, vthp, amask,
                                                              opart, mlp);
  attn_combine<<<B_ * NH_ * K_, blk, 0, stream>>>(opart, mlp, attnw);
  gemm_xwT<<<dim3(2048 / 64, 512 / 64), blk, 0, stream>>>(attnw, wo, (float*)d_out,
                                                          512, 2048, 2048);
}
